// TransformerLayer_12343736009121
// MI455X (gfx1250) — matrix operand order repaired and checked by reading, not compiled, not run on hardware
//
#include <hip/hip_runtime.h>
#include <hip/hip_bf16.h>

typedef __attribute__((ext_vector_type(16))) _Float16 v16h;
typedef __attribute__((ext_vector_type(8)))  _Float16 v8h;
typedef __attribute__((ext_vector_type(8)))  float    v8f;

#define FDIM  256
#define BATCH 16
#define NSEQ  2048

// ---------------------------------------------------------------------------
// WMMA helpers (CDNA5 wave32, V_WMMA_F32_16X16X32_F16)
// ---------------------------------------------------------------------------
__device__ __forceinline__ v8f wmma_f16(v16h a, v16h b, v8f c) {
  // 8 args: (neg_a, A, neg_b, B, c_mod, C, reuse_a, reuse_b)
  return __builtin_amdgcn_wmma_f32_16x16x32_f16(false, a, false, b, (short)0, c,
                                                false, false);
}

__device__ __forceinline__ v16h cat8(v8h lo, v8h hi) {
  return __builtin_shufflevector(lo, hi, 0, 1, 2, 3, 4, 5, 6, 7,
                                 8, 9, 10, 11, 12, 13, 14, 15);
}

// A-operand (16x32 f16, M=row, K=f), contiguous k order (matches B packing):
//   lane L: m = L%16 ; g = L/16
//   halves 0..15 -> k = fb*32 + 16g + h
// => one contiguous 32B chunk per lane from a row-major row.
__device__ __forceinline__ v16h load_a_tile(const _Float16* row, int fb, int lg) {
  return *(const v16h*)(row + fb * 32 + 16 * lg);
}

// Async copy of one 16KB (32 x 256 f16) tile global -> LDS, row-major.
// GVS mode: mem_addr = SADDR(SGPR64) + VADDR(VGPR32) ; VDST = LDS byte addr.
// 1024 x 16B transfers spread over 128 threads (8 per thread), ASYNCcnt.
__device__ __forceinline__ void async_copy_tile_16k(const _Float16* __restrict__ src,
                                                    const _Float16* dst_lds,
                                                    int tid) {
  unsigned base = (unsigned)(size_t)dst_lds;  // low 32 bits = LDS byte offset
#pragma unroll
  for (int j = 0; j < 8; ++j) {
    unsigned off  = (unsigned)((j * 128 + tid) * 16);
    unsigned ldsa = base + off;
    asm volatile("global_load_async_to_lds_b128 %0, %1, %2"
                 :: "v"(ldsa), "v"(off), "s"(src)
                 : "memory");
  }
}

// ---------------------------------------------------------------------------
// Kernel 1: pre-swizzle f32 weights into WMMA B-operand-packed f16 layout.
// B-operand (32x16, K=f_in, N=f_out): lane L holds n = L%16, k = 16*(L/16)+h.
// Stored value = W[n_global][k_global] (since y = h @ W^T).
// Layout: wpack[(mat*8+kb)*16+cb][lane][h], 32B contiguous per lane.
// mats: 0=W2 (pos), 1=Wk (query! ref swaps), 2=Wq (key), 3=Wv.
// ---------------------------------------------------------------------------
__global__ __launch_bounds__(256) void pack_weights_kernel(
    const float* __restrict__ Wq, const float* __restrict__ Wk,
    const float* __restrict__ Wv, const float* __restrict__ W2,
    _Float16* __restrict__ wpack) {
  int tid  = blockIdx.x * 256 + threadIdx.x;  // 0..16383
  int lane = tid & 31;
  int cb   = (tid >> 5) & 15;
  int kb   = (tid >> 9) & 7;
  int mat  = tid >> 12;
  const float* W = (mat == 0) ? W2 : (mat == 1) ? Wk : (mat == 2) ? Wq : Wv;
  int n  = cb * 16 + (lane & 15);
  int k0 = kb * 32 + (lane >> 4) * 16;
  _Float16* dst = wpack + ((size_t)(mat * 8 + kb) * 16 + cb) * 512 + lane * 16;
#pragma unroll
  for (int h = 0; h < 16; ++h) dst[h] = (_Float16)W[n * FDIM + k0 + h];
}

// ---------------------------------------------------------------------------
// Kernel 2: pos-encoder + h = x + pos, then q/k/v projections (all WMMA).
// One WG = 128 threads (4 waves), 64 rows of one batch; wave owns 16 rows.
// ---------------------------------------------------------------------------
__global__ __launch_bounds__(128) void qkv_proj_kernel(
    const float* __restrict__ x, const float* __restrict__ p,
    const float* __restrict__ W1, const float* __restrict__ b1,
    const float* __restrict__ b2, const float* __restrict__ bias_q,
    const float* __restrict__ bias_k, const float* __restrict__ bias_v,
    const _Float16* __restrict__ wpack, _Float16* __restrict__ qout,
    _Float16* __restrict__ kout, _Float16* __restrict__ vout) {
  __shared__ __align__(32) _Float16 t_lds[64 * FDIM];  // 32 KB
  __shared__ __align__(32) _Float16 h_lds[64 * FDIM];  // 32 KB

  const int tid  = threadIdx.x;
  const int lane = tid & 31;
  const int wave = tid >> 5;
  const int ln   = lane & 15;
  const int lg   = lane >> 4;
  const int bb   = blockIdx.x >> 5;          // batch
  const int row0 = (blockIdx.x & 31) * 64;   // row block

  // --- stage 1: t = relu(p @ W1^T + b1), f16 into LDS (tiny 3-wide GEMM) ---
  const float* pb = p + ((size_t)bb * NSEQ + row0) * 3;
  for (int i = tid; i < 64 * FDIM; i += 128) {
    int row = i >> 8, f = i & 255;
    float acc = b1[f] + pb[row * 3 + 0] * W1[f * 3 + 0] +
                pb[row * 3 + 1] * W1[f * 3 + 1] +
                pb[row * 3 + 2] * W1[f * 3 + 2];
    t_lds[row * FDIM + f] = (_Float16)fmaxf(acc, 0.0f);
  }
  __syncthreads();

  // --- stage 2: pos = t @ W2^T + b2 ; h = x + pos -> f16 LDS ---
  const _Float16* trow = t_lds + (wave * 16 + ln) * FDIM;
  const float* xrow = x + ((size_t)bb * NSEQ + row0 + wave * 16) * FDIM;
#pragma unroll 1
  for (int cb = 0; cb < 16; ++cb) {
    v8f acc = {};
#pragma unroll
    for (int fb = 0; fb < 8; ++fb) {
      v16h a = load_a_tile(trow, fb, lg);
      v16h b = *(const v16h*)(wpack + ((size_t)((0 * 8 + fb) * 16 + cb)) * 512 +
                              lane * 16);
      acc = wmma_f16(a, b, acc);
    }
    float bias = b2[cb * 16 + ln];
#pragma unroll
    for (int r = 0; r < 8; ++r) {  // C layout: m = r + 8g, n = L%16
      int m = r + 8 * lg;
      float hv = acc[r] + bias + xrow[(size_t)m * FDIM + cb * 16 + ln];
      h_lds[(wave * 16 + m) * FDIM + cb * 16 + ln] = (_Float16)hv;
    }
  }
  __syncthreads();

  // --- stage 3: q = h@Wk^T+bk, k = h@Wq^T+bq, v = h@Wv^T+bv (ref swap) ---
  const _Float16* hrow = h_lds + (wave * 16 + ln) * FDIM;
#pragma unroll 1
  for (int proj = 0; proj < 3; ++proj) {
    const float* bias_p = (proj == 0) ? bias_q : (proj == 1) ? bias_k : bias_v;
    _Float16* dst = (proj == 0) ? qout : (proj == 1) ? kout : vout;
    _Float16* drow = dst + ((size_t)bb * NSEQ + row0 + wave * 16) * FDIM;
#pragma unroll 1
    for (int cb = 0; cb < 16; ++cb) {
      v8f acc = {};
#pragma unroll
      for (int fb = 0; fb < 8; ++fb) {
        v16h a = load_a_tile(hrow, fb, lg);
        v16h b = *(const v16h*)(wpack +
                                ((size_t)(((proj + 1) * 8 + fb) * 16 + cb)) * 512 +
                                lane * 16);
        acc = wmma_f16(a, b, acc);
      }
      float bias = bias_p[cb * 16 + ln];
#pragma unroll
      for (int r = 0; r < 8; ++r) {
        int m = r + 8 * lg;
        drow[(size_t)m * FDIM + cb * 16 + ln] = (_Float16)(acc[r] + bias);
      }
    }
  }
}

// ---------------------------------------------------------------------------
// Kernel 3: flash attention. WG = 128 threads (4 waves) = 64 query rows.
// K tiles are double-buffered in LDS via async global->LDS DMA (ASYNCcnt),
// so tile kt+1 streams in while tile kt feeds the matrix cores.
// ---------------------------------------------------------------------------
__global__ __launch_bounds__(128) void flash_attn_kernel(
    const _Float16* __restrict__ qbuf, const _Float16* __restrict__ kbuf,
    const _Float16* __restrict__ vbuf, float* __restrict__ out) {
  __shared__ __align__(32) _Float16 k_lds[2][32 * FDIM];  // 2 x 16 KB, row-major
  __shared__ __align__(32) _Float16 v_lds[16 * 32 * 16];  // 16 KB, B-op packed
  __shared__ __align__(32) _Float16 p_lds[4][16 * 32];    // 4 KB, per-wave P

  const int tid  = threadIdx.x;
  const int lane = tid & 31;
  const int wave = tid >> 5;
  const int ln   = lane & 15;
  const int lg   = lane >> 4;
  const int bb   = blockIdx.x >> 5;
  const int q0   = (blockIdx.x & 31) * 64;

  // Q A-operands are invariant over the key loop: hoist into registers.
  const _Float16* qrow =
      qbuf + ((size_t)bb * NSEQ + q0 + wave * 16 + ln) * FDIM;
  v16h aq[8];
#pragma unroll
  for (int fb = 0; fb < 8; ++fb) aq[fb] = load_a_tile(qrow, fb, lg);

  v8f accO[16];
#pragma unroll
  for (int cb = 0; cb < 16; ++cb) accO[cb] = (v8f){};
  float Mrun[8], Lrun[8];
#pragma unroll
  for (int r = 0; r < 8; ++r) { Mrun[r] = -3.0e38f; Lrun[r] = 0.0f; }

  const _Float16* kbb = kbuf + (size_t)bb * NSEQ * FDIM;
  const _Float16* vbb = vbuf + (size_t)bb * NSEQ * FDIM;

  // Prime the async pipeline with key tile 0.
  async_copy_tile_16k(kbb, &k_lds[0][0], tid);

  for (int kt = 0; kt < NSEQ / 32; ++kt) {
    const int key0 = kt * 32;
    const int buf  = kt & 1;

    // Tile kt landed (our ASYNCcnt) ...
    asm volatile("s_wait_asynccnt 0x0" ::: "memory");
    // ... and every wave is past its reads of k_lds[buf^1] / v_lds.
    __syncthreads();

    // Kick off the DMA for tile kt+1 into the other buffer.
    if (kt + 1 < NSEQ / 32)
      async_copy_tile_16k(kbb + (size_t)(key0 + 32) * FDIM, &k_lds[buf ^ 1][0],
                          tid);

    // Stage V tile (32 keys x 256 f) into B-operand-packed layout:
    // v_lds[(cb*32 + lane)*16 + h] = V[key0 + 16*(lane/16) + h][cb*16 + lane%16]
    for (int i = tid; i < 1024; i += 128) {
      int kr = i >> 5;
      int f0 = (i & 31) * 8;
      v8h vv = *(const v8h*)(vbb + (size_t)(key0 + kr) * FDIM + f0);
#pragma unroll
      for (int j = 0; j < 8; ++j) {
        int f = f0 + j;
        int cb = f >> 4;
        int lnn = (f & 15) + ((kr >> 4) << 4);
        int h = kr & 15;
        v_lds[(cb * 32 + lnn) * 16 + h] = vv[j];
      }
    }
    __syncthreads();

    // S = Q @ K^T for this wave's 16 q rows x 32 keys (two 16x16 tiles).
    // K B-operand from LDS: lane L -> n = key L%16, k = fb*32 + 16*(L/16) + h
    // => 32 contiguous, 32B-aligned bytes per lane (ds_load_b128 pairs).
    const _Float16* kl = &k_lds[buf][0];
    v8f S0 = {}, S1 = {};
#pragma unroll
    for (int fb = 0; fb < 8; ++fb) {
      v16h b0 = *(const v16h*)(kl + (size_t)ln * FDIM + fb * 32 + 16 * lg);
      S0 = wmma_f16(aq[fb], b0, S0);
    }
#pragma unroll
    for (int fb = 0; fb < 8; ++fb) {
      v16h b1 = *(const v16h*)(kl + (size_t)(16 + ln) * FDIM + fb * 32 + 16 * lg);
      S1 = wmma_f16(aq[fb], b1, S1);
    }

    // Online softmax. Row m = r + 8g lives across the 16 lanes of one half,
    // so row reductions are shfl_xor over masks 8,4,2,1.
#pragma unroll
    for (int r = 0; r < 8; ++r) {
      float s0 = S0[r] * 0.0625f;  // 1/sqrt(256)
      float s1 = S1[r] * 0.0625f;
      float rmax = fmaxf(s0, s1);
      rmax = fmaxf(rmax, __shfl_xor(rmax, 8, 32));
      rmax = fmaxf(rmax, __shfl_xor(rmax, 4, 32));
      rmax = fmaxf(rmax, __shfl_xor(rmax, 2, 32));
      rmax = fmaxf(rmax, __shfl_xor(rmax, 1, 32));
      float Mnew  = fmaxf(Mrun[r], rmax);
      float alpha = __expf(Mrun[r] - Mnew);
      Mrun[r] = Mnew;
      float p0 = __expf(s0 - Mnew);
      float p1 = __expf(s1 - Mnew);
      float rs = p0 + p1;
      rs += __shfl_xor(rs, 8, 32);
      rs += __shfl_xor(rs, 4, 32);
      rs += __shfl_xor(rs, 2, 32);
      rs += __shfl_xor(rs, 1, 32);
      Lrun[r] = Lrun[r] * alpha + rs;
#pragma unroll
      for (int cb = 0; cb < 16; ++cb) accO[cb][r] *= alpha;
      int m = r + 8 * lg;  // route P through per-wave LDS: C-layout -> A-layout
      p_lds[wave][m * 32 + ln]      = (_Float16)p0;
      p_lds[wave][m * 32 + 16 + ln] = (_Float16)p1;
    }
    asm volatile("s_wait_dscnt 0x0" ::: "memory");  // wave-local LDS RAW

    // P A-operand (16x32, K = key_local), contiguous k = 16g + h.
    v16h ap = *(const v16h*)(&p_lds[wave][ln * 32 + 16 * lg]);

    // O += P @ V : one WMMA per 16-col tile, B straight from packed LDS.
#pragma unroll
    for (int cb = 0; cb < 16; ++cb) {
      v16h bv = *(const v16h*)(&v_lds[(cb * 32 + lane) * 16]);
      accO[cb] = wmma_f16(ap, bv, accO[cb]);
    }
  }

  // Normalize and write out (f32, C-layout scatter).
  float* orow = out + ((size_t)bb * NSEQ + q0 + wave * 16) * FDIM;
#pragma unroll
  for (int cb = 0; cb < 16; ++cb) {
#pragma unroll
    for (int r = 0; r < 8; ++r) {
      int m = r + 8 * lg;
      orow[(size_t)m * FDIM + cb * 16 + ln] = accO[cb][r] / Lrun[r];
    }
  }
}

// ---------------------------------------------------------------------------
// Host launcher. Workspace: 512 KB packed weights + 3 x 16.8 MB f16 q/k/v.
// ---------------------------------------------------------------------------
extern "C" void kernel_launch(void* const* d_in, const int* in_sizes, int n_in,
                              void* d_out, int out_size, void* d_ws,
                              size_t ws_size, hipStream_t stream) {
  const float* x  = (const float*)d_in[0];
  const float* p  = (const float*)d_in[1];
  const float* Wq = (const float*)d_in[2];
  const float* bq = (const float*)d_in[3];
  const float* Wk = (const float*)d_in[4];
  const float* bk = (const float*)d_in[5];
  const float* Wv = (const float*)d_in[6];
  const float* bv = (const float*)d_in[7];
  const float* W1 = (const float*)d_in[8];
  const float* b1 = (const float*)d_in[9];
  const float* W2 = (const float*)d_in[10];
  const float* b2 = (const float*)d_in[11];
  float* out = (float*)d_out;

  _Float16* wpack = (_Float16*)d_ws;            // 4*8*16*32*16 = 262144 halves
  const size_t qkv = (size_t)BATCH * NSEQ * FDIM;
  _Float16* qbuf = wpack + 4 * 8 * 16 * 32 * 16;
  _Float16* kbuf = qbuf + qkv;
  _Float16* vbuf = kbuf + qkv;

  pack_weights_kernel<<<64, 256, 0, stream>>>(Wq, Wk, Wv, W2, wpack);
  // NOTE the reference name swap: q uses (Wk,bk); k uses (Wq,bq).
  qkv_proj_kernel<<<BATCH * (NSEQ / 64), 128, 0, stream>>>(
      x, p, W1, b1, b2, /*bias_q=*/bk, /*bias_k=*/bq, /*bias_v=*/bv, wpack,
      qbuf, kbuf, vbuf);
  flash_attn_kernel<<<BATCH * (NSEQ / 64), 128, 0, stream>>>(qbuf, kbuf, vbuf,
                                                             out);
}
